// SlotAttention_80556406604615
// MI455X (gfx1250) — matrix operand order repaired and checked by reading, not compiled, not run on hardware
//
#include <hip/hip_runtime.h>

// ---------------------------------------------------------------------------
// Slot Attention, MI455X (gfx1250, wave32).
// Memory-bound problem: single fused pass over inputs (134 MB), k/v cached in
// L2 as bf16 (128 MB < 192 MB L2), all big GEMMs via v_wmma_f32_16x16x32_bf16.
// Round 4: async global<->LDS movers with the correct builtin pointer types
// (int4 in AS(1)/AS(3), per hipcc's diagnostic from round 3).
// ---------------------------------------------------------------------------

typedef __attribute__((ext_vector_type(16))) __bf16 v16bf;
typedef __attribute__((ext_vector_type(8)))  __bf16 v8bf;
typedef __attribute__((ext_vector_type(8)))  float  v8f;

#define WMMA_BF16(A, B, C) \
  __builtin_amdgcn_wmma_f32_16x16x32_bf16(false, (A), false, (B), (short)0, (C), false, false)

#if __has_builtin(__builtin_amdgcn_global_load_async_to_lds_b128) && \
    __has_builtin(__builtin_amdgcn_s_wait_asynccnt)
#define HAVE_ASYNC_LD 1
#else
#define HAVE_ASYNC_LD 0
#endif
#if HAVE_ASYNC_LD && __has_builtin(__builtin_amdgcn_global_store_async_from_lds_b128)
#define HAVE_ASYNC_ST 1
#else
#define HAVE_ASYNC_ST 0
#endif

// Builtin parameter types per hipcc diagnostic:
//   'int __attribute__((__vector_size__(4*sizeof(int)))) __device__ *'
typedef int v4i __attribute__((vector_size(16)));
typedef __attribute__((address_space(1))) v4i gv4i;   // global (AS1)
typedef __attribute__((address_space(3))) v4i lv4i;   // LDS    (AS3)

// flat->AS1 is bit-identical for global memory; flat LDS addr low 32 bits are
// the LDS byte address (ISA 10.2 aperture rules).
__device__ __forceinline__ gv4i* as_global_v4(const void* p) {
  return (gv4i*)(unsigned long long)(size_t)p;
}
__device__ __forceinline__ lv4i* as_lds_v4(const void* p) {
  return (lv4i*)(unsigned)(size_t)p;
}

__device__ __forceinline__ v16bf cat16(v8bf lo, v8bf hi) {
  v16bf r;
#pragma unroll
  for (int i = 0; i < 8; ++i) { r[i] = lo[i]; r[i + 8] = hi[i]; }
  return r;
}

__device__ __forceinline__ float block64_sum(float v, float* red, int tid) {
  __syncthreads();
  red[tid] = v;
  __syncthreads();
#pragma unroll
  for (int o = 32; o > 0; o >>= 1) {
    if (tid < o) red[tid] += red[tid + o];
    __syncthreads();
  }
  float r = red[0];
  __syncthreads();
  return r;
}

// --------------------------- slots init ------------------------------------
__global__ void init_slots_kernel(const float* __restrict__ eps,
                                  const float* __restrict__ mu,
                                  const float* __restrict__ logvar,
                                  float* __restrict__ slots, int n) {
  int i = blockIdx.x * blockDim.x + threadIdx.x;
  if (i < n) {
    int d = i & 63;
    slots[i] = mu[d] + __expf(0.5f * logvar[d]) * eps[i];
  }
}

// ------- one-shot: pre-swizzle k_w / v_w into bf16 WMMA B-fragments --------
// wfrag[m][nt][t][lane][j] = W_m[nt*16 + (lane&15)][32*t + 8*(lane>>4) + (j&7) + 16*(j>>3)]
// (k_w scaled by 1/sqrt(D)=0.125).  8192 elements total = 16 KB.
__global__ __launch_bounds__(256) void prep_wfrag_kernel(
    const float* __restrict__ kw, const float* __restrict__ vw,
    __bf16* __restrict__ wfrag) {
  const int i = blockIdx.x * blockDim.x + threadIdx.x;  // 0..8191
  const int j    = i & 15;
  const int lane = (i >> 4) & 31;
  const int t    = (i >> 9) & 1;
  const int nt   = (i >> 10) & 3;
  const int m    = (i >> 12) & 1;
  const int d = nt * 16 + (lane & 15);
  const int c = 32 * t + 8 * (lane >> 4) + (j & 7) + 16 * (j >> 3);
  const float scale = m ? 1.f : 0.125f;
  const float* W = m ? vw : kw;
  wfrag[i] = (__bf16)(W[d * 64 + c] * scale);
}

// --------------- fused LayerNorm + k/v projection (bf16 out) ---------------
// grid: 4096 blocks x 256 threads; each wave handles a 16x64 row tile.
__global__ __launch_bounds__(256) void lnproj_kernel(
    const float* __restrict__ x,
    const __bf16* __restrict__ wfrag,
    const float* __restrict__ niw, const float* __restrict__ nib,
    __bf16* __restrict__ kbf, __bf16* __restrict__ vbf) {
  __shared__ __align__(16) __bf16 xn[8][16][64];
  __shared__ __align__(16) __bf16 stg[8][16][64];
  const int tid = threadIdx.x, wave = tid >> 5, lane = tid & 31;
  const int row = lane & 15, khalf = lane >> 4;
  const long rowbase = (long)blockIdx.x * 128 + wave * 16;

  // ---- load half a row per lane, LayerNorm (lane pair l,l+16 per row) ----
  const float* xr = x + (rowbase + row) * 64 + khalf * 32;
  float xv[32];
  float s = 0.f, s2 = 0.f;
#pragma unroll
  for (int i = 0; i < 8; ++i) {
    float4 f = ((const float4*)xr)[i];
    xv[4 * i + 0] = f.x; xv[4 * i + 1] = f.y;
    xv[4 * i + 2] = f.z; xv[4 * i + 3] = f.w;
    s  += f.x + f.y + f.z + f.w;
    s2 += f.x * f.x + f.y * f.y + f.z * f.z + f.w * f.w;
  }
  s  += __shfl_xor(s, 16);
  s2 += __shfl_xor(s2, 16);
  const float mean = s * (1.f / 64.f);
  const float var  = s2 * (1.f / 64.f) - mean * mean;
  const float rstd = rsqrtf(var + 1e-5f);
#pragma unroll
  for (int i = 0; i < 32; ++i) {
    const int c = khalf * 32 + i;
    xn[wave][row][c] = (__bf16)((xv[i] - mean) * rstd * niw[c] + nib[c]);
  }
  __syncthreads();

  // ---- A fragments (16x32 bf16, two K-tiles covering Din=64) ----
  const __bf16* xp = &xn[wave][row][0];
  const v16bf a0 = cat16(*(const v8bf*)(xp + 8 * khalf),
                         *(const v8bf*)(xp + 16 + 8 * khalf));
  const v16bf a1 = cat16(*(const v8bf*)(xp + 32 + 8 * khalf),
                         *(const v8bf*)(xp + 48 + 8 * khalf));

#pragma unroll
  for (int m = 0; m < 2; ++m) {
    __bf16* dst = (m ? vbf : kbf) + rowbase * 64;
#pragma unroll
    for (int nt = 0; nt < 4; ++nt) {
      // pre-swizzled B fragments: direct 32-byte loads, no conversion
      const v16bf b0 = *(const v16bf*)(
          wfrag + ((((size_t)m * 4 + nt) * 2 + 0) * 32 + lane) * 16);
      const v16bf b1 = *(const v16bf*)(
          wfrag + ((((size_t)m * 4 + nt) * 2 + 1) * 32 + lane) * 16);
      v8f c = {};
      c = WMMA_BF16(a0, b0, c);
      c = WMMA_BF16(a1, b1, c);
#pragma unroll
      for (int rr = 0; rr < 8; ++rr)
        stg[wave][khalf * 8 + rr][nt * 16 + row] = (__bf16)c[rr];
    }
    __syncthreads();
#if HAVE_ASYNC_ST
    {
      // stg tile == row-major global tile: stream LDS -> global (ASYNCcnt)
      __bf16* lsrc = &stg[wave][0][0] + lane * 32;  // 64 B per lane
      __bf16* gdst = dst + lane * 32;
#pragma unroll
      for (int i = 0; i < 4; ++i)
        __builtin_amdgcn_global_store_async_from_lds_b128(
            as_global_v4(gdst + i * 8), as_lds_v4(lsrc + i * 8), 0, 0);
      __builtin_amdgcn_s_wait_asynccnt(0);
    }
#else
    {
      const unsigned int* sp = (const unsigned int*)&stg[wave][0][0];
      unsigned int* dp = (unsigned int*)dst;
#pragma unroll
      for (int rr = 0; rr < 16; ++rr) dp[rr * 32 + lane] = sp[rr * 32 + lane];
    }
#endif
    __syncthreads();
  }
}

// --------- q = LN(slots) @ q_w^T, written in pre-swizzled B-fragment -------
// grid: 1024 blocks (b*8+s) x 64 threads.
__global__ __launch_bounds__(64) void q_kernel(
    const float* __restrict__ slots, const float* __restrict__ qw,
    const float* __restrict__ nsw, const float* __restrict__ nsb,
    __bf16* __restrict__ qfrag) {
  __shared__ float xn[64];
  __shared__ float red[64];
  const int tid = threadIdx.x;
  const int b = blockIdx.x >> 3, s = blockIdx.x & 7;
  const float v = slots[(size_t)blockIdx.x * 64 + tid];
  float mean = block64_sum(v, red, tid) * (1.f / 64.f);
  float dv = v - mean;
  float var = block64_sum(dv * dv, red, tid) * (1.f / 64.f);
  float rstd = rsqrtf(var + 1e-5f);
  xn[tid] = dv * rstd * nsw[tid] + nsb[tid];
  __syncthreads();
  float acc = 0.f;
  const float* wr = qw + (size_t)tid * 64;
#pragma unroll 8
  for (int c = 0; c < 64; ++c) acc += xn[c] * wr[c];
  // qfrag[b][t][lane][j] = q[n=lane&15][k=32t+8*(lane>>4)+(j&7)+16*(j>>3)], cols 8..15 = 0
  const int t = tid >> 5, kh = (tid >> 3) & 1, j = (tid & 7) + 8 * ((tid >> 4) & 1);
  __bf16* base = qfrag + ((size_t)(b * 2 + t) * 32) * 16;
  base[(kh * 16 + s) * 16 + j]     = (__bf16)acc;
  base[(kh * 16 + s + 8) * 16 + j] = (__bf16)0.f;
}

// ------------- logits + softmax + colsum + w^T·v accumulation --------------
// grid: 512 blocks (b*4 + ntile) x 256 threads; wave = 128 tokens (4x32).
__global__ __launch_bounds__(256) void attn_kernel(
    const __bf16* __restrict__ kbf, const __bf16* __restrict__ vbf,
    const __bf16* __restrict__ qfrag, float* __restrict__ attn_raw,
    float* __restrict__ colsum, float* __restrict__ w_out, int store_w) {
  __shared__ __align__(16) __bf16 wt[8][32][16];
  __shared__ __align__(16) __bf16 vt[8][32][64];
  const int tid = threadIdx.x, wave = tid >> 5, lane = tid & 31;
  const int col = lane & 15, khalf = lane >> 4;
  const int b = blockIdx.x >> 2;
  const long n0 = ((long)b << 12) + (long)(blockIdx.x & 3) * 1024 + wave * 128;

  const v16bf q0 = *(const v16bf*)(qfrag + (((size_t)b * 2 + 0) * 32 + lane) * 16);
  const v16bf q1 = *(const v16bf*)(qfrag + (((size_t)b * 2 + 1) * 32 + lane) * 16);

  v8f acc[4] = {};
  float csum = 0.f;

  // warm the near caches for the first chunk (one 128B row per lane)
  __builtin_prefetch(kbf + (n0 + lane) * 64, 0, 3);
  __builtin_prefetch(vbf + (n0 + lane) * 64, 0, 3);

  for (int ch = 0; ch < 4; ++ch) {
    const long tb = n0 + ch * 32;
    // prefetch next chunk's k/v rows (global_prefetch_b8) while we compute
    if (ch < 3) {
      __builtin_prefetch(kbf + (tb + 32 + lane) * 64, 0, 3);
      __builtin_prefetch(vbf + (tb + 32 + lane) * 64, 0, 3);
    }
#if HAVE_ASYNC_LD
    // ---- async-stage V tile global->LDS (no VGPR round-trip, ASYNCcnt) ----
    {
      const __bf16* src = vbf + (tb + lane) * 64;
      __bf16* dst = &vt[wave][lane][0];
#pragma unroll
      for (int i = 0; i < 8; ++i)
        __builtin_amdgcn_global_load_async_to_lds_b128(
            as_global_v4(src + i * 8), as_lds_v4(dst + i * 8), 0, 0);
    }
#endif
    // ---- logits (k·qT) + softmax over slots, two 16-token subtiles ----
#pragma unroll
    for (int st = 0; st < 2; ++st) {
      const __bf16* kr = kbf + (tb + st * 16 + col) * 64;
      v16bf a0 = cat16(*(const v8bf*)(kr + 8 * khalf),
                       *(const v8bf*)(kr + 16 + 8 * khalf));
      v16bf a1 = cat16(*(const v8bf*)(kr + 32 + 8 * khalf),
                       *(const v8bf*)(kr + 48 + 8 * khalf));
      v8f c = {};
      c = WMMA_BF16(a0, q0, c);
      c = WMMA_BF16(a1, q1, c);
#pragma unroll
      for (int r = 0; r < 8; ++r) {
        float l = c[r];
        float mx = l;
        mx = fmaxf(mx, __shfl_xor(mx, 1));
        mx = fmaxf(mx, __shfl_xor(mx, 2));
        mx = fmaxf(mx, __shfl_xor(mx, 4));
        float e = __expf(l - mx);
        float ss = e;
        ss += __shfl_xor(ss, 1);
        ss += __shfl_xor(ss, 2);
        ss += __shfl_xor(ss, 4);
        float w = e / ss + 1e-8f;          // softmax + EPS (unnormalized)
        csum += w;
        const int tloc = st * 16 + khalf * 8 + r;
        wt[wave][tloc][col] = (__bf16)w;
        if (store_w && col < 8) w_out[(tb + tloc) * 8 + col] = w;
      }
    }
#if HAVE_ASYNC_LD
    __builtin_amdgcn_s_wait_asynccnt(0);   // V tile resident in LDS
#else
    // ---- stage V tile (one 128-byte row per lane) ----
    {
      const float4* src = (const float4*)(vbf + (tb + lane) * 64);
      float4* dst = (float4*)(&vt[wave][lane][0]);
#pragma unroll
      for (int i = 0; i < 8; ++i) dst[i] = src[i];
    }
#endif
    __syncthreads();
    // ---- A fragment of W^T (M=slot, K=32 tokens) ----
    v16bf aw;
#pragma unroll
    for (int j = 0; j < 8; ++j) {
      aw[j]     = wt[wave][khalf * 8 + j][col];
      aw[j + 8] = wt[wave][16 + khalf * 8 + j][col];
    }
    // ---- atten accumulation: acc[nt] += W^T x V ----
#pragma unroll
    for (int nt = 0; nt < 4; ++nt) {
      v16bf bv;
#pragma unroll
      for (int j = 0; j < 16; ++j) bv[j] = vt[wave][khalf * 8 + (j & 7) + 16 * (j >> 3)][nt * 16 + col];
      acc[nt] = WMMA_BF16(aw, bv, acc[nt]);
    }
    __syncthreads();
  }
  // ---- reductions to global ----
  csum += __shfl_xor(csum, 16);
  if (lane < 8) atomicAdd(&colsum[b * 16 + lane], csum);
#pragma unroll
  for (int nt = 0; nt < 4; ++nt)
#pragma unroll
    for (int r = 0; r < 8; ++r)
      atomicAdd(&attn_raw[(b * 16 + khalf * 8 + r) * 64 + nt * 16 + col],
                acc[nt][r]);
}

// -------------------- GRU cell + residual MLP (1024 rows) ------------------
__global__ __launch_bounds__(64) void gru_mlp_kernel(
    float* __restrict__ slots, const float* __restrict__ attn_raw,
    const float* __restrict__ colsum,
    const float* __restrict__ wih, const float* __restrict__ whh,
    const float* __restrict__ bih, const float* __restrict__ bhh,
    const float* __restrict__ nrw, const float* __restrict__ nrb,
    const float* __restrict__ w1, const float* __restrict__ w2,
    const float* __restrict__ b2) {
  __shared__ float a[64], hp[64], ln[64], tt[128];
  __shared__ float red[64];
  const int tid = threadIdx.x;
  const int b = blockIdx.x >> 3, s = blockIdx.x & 7;
  const float cs = colsum[b * 16 + s];
  const float av = attn_raw[(size_t)(b * 16 + s) * 64 + tid] / cs;
  const float hold = slots[(size_t)blockIdx.x * 64 + tid];
  a[tid] = av;
  hp[tid] = hold;
  __syncthreads();
  float gsum[3], hsum[3];
#pragma unroll
  for (int g = 0; g < 3; ++g) {
    const float* wi = wih + (size_t)(g * 64 + tid) * 64;
    const float* wh = whh + (size_t)(g * 64 + tid) * 64;
    float si = bih[g * 64 + tid], sh = bhh[g * 64 + tid];
    for (int c = 0; c < 64; ++c) { si += a[c] * wi[c]; sh += hp[c] * wh[c]; }
    gsum[g] = si; hsum[g] = sh;
  }
  const float r = 1.f / (1.f + __expf(-(gsum[0] + hsum[0])));
  const float z = 1.f / (1.f + __expf(-(gsum[1] + hsum[1])));
  const float nn = tanhf(gsum[2] + r * hsum[2]);
  const float h = (1.f - z) * nn + z * hold;
  // LayerNorm(h)
  float mean = block64_sum(h, red, tid) * (1.f / 64.f);
  float dv = h - mean;
  float var = block64_sum(dv * dv, red, tid) * (1.f / 64.f);
  float rstd = rsqrtf(var + 1e-5f);
  ln[tid] = dv * rstd * nrw[tid] + nrb[tid];
  __syncthreads();
  // hidden = relu(ln @ w1^T)
#pragma unroll
  for (int half = 0; half < 2; ++half) {
    const int jj = half * 64 + tid;
    const float* wr = w1 + (size_t)jj * 64;
    float acc = 0.f;
    for (int c = 0; c < 64; ++c) acc += ln[c] * wr[c];
    tt[jj] = fmaxf(acc, 0.f);
  }
  __syncthreads();
  float out = h + b2[tid];
  const float* wr2 = w2 + (size_t)tid * 128;
  for (int j = 0; j < 128; ++j) out += tt[j] * wr2[j];
  slots[(size_t)blockIdx.x * 64 + tid] = out;
}

// ------------------- final w normalization (over tokens) -------------------
__global__ void normalize_w_kernel(float* __restrict__ w,
                                   const float* __restrict__ colsum, int n) {
  int i = blockIdx.x * blockDim.x + threadIdx.x;
  if (i < n) {
    int s = i & 7;
    int b = i >> 15;  // 4096 tokens * 8 slots per batch
    w[i] = w[i] / colsum[b * 16 + s];
  }
}

// ---------------------------------------------------------------------------
extern "C" void kernel_launch(void* const* d_in, const int* in_sizes, int n_in,
                              void* d_out, int out_size, void* d_ws,
                              size_t ws_size, hipStream_t stream) {
  const float* inputs  = (const float*)d_in[0];
  const float* eps     = (const float*)d_in[1];
  const float* mu      = (const float*)d_in[2];
  const float* logvar  = (const float*)d_in[3];
  const float* kw      = (const float*)d_in[4];
  const float* vw      = (const float*)d_in[5];
  const float* qw      = (const float*)d_in[6];
  const float* niw     = (const float*)d_in[7];
  const float* nib     = (const float*)d_in[8];
  const float* nsw     = (const float*)d_in[9];
  const float* nsb     = (const float*)d_in[10];
  const float* nrw     = (const float*)d_in[11];
  const float* nrb     = (const float*)d_in[12];
  const float* wih     = (const float*)d_in[13];
  const float* whh     = (const float*)d_in[14];
  const float* bih     = (const float*)d_in[15];
  const float* bhh     = (const float*)d_in[16];
  const float* w1      = (const float*)d_in[17];
  const float* w2      = (const float*)d_in[18];
  const float* b2      = (const float*)d_in[19];

  float* slots = (float*)d_out;                 // [128,8,64]
  float* w_out = slots + 128 * 8 * 64;          // [128,4096,8]

  char* ws = (char*)d_ws;
  __bf16* kbf   = (__bf16*)ws;                                   // 64 MB
  __bf16* vbf   = (__bf16*)(ws + 67108864ull);                   // 64 MB
  __bf16* qfrag = (__bf16*)(ws + 134217728ull);                  // 256 KB
  float*  attn_raw = (float*)(ws + 134479872ull);                // [128,16,64]
  float*  colsum   = attn_raw + 128 * 16 * 64;                   // [128,16]
  __bf16* wfrag = (__bf16*)(ws + 135012352ull);                  // 16 KB
  const size_t acc_bytes = (size_t)(128 * 16 * 64 + 128 * 16) * sizeof(float);

  init_slots_kernel<<<(128 * 8 * 64) / 256, 256, 0, stream>>>(
      eps, mu, logvar, slots, 128 * 8 * 64);

  prep_wfrag_kernel<<<32, 256, 0, stream>>>(kw, vw, wfrag);

  lnproj_kernel<<<4096, 256, 0, stream>>>(inputs, wfrag, niw, nib, kbf, vbf);

  for (int it = 0; it < 4; ++it) {
    q_kernel<<<1024, 64, 0, stream>>>(slots, qw, nsw, nsb, qfrag);
    (void)hipMemsetAsync(attn_raw, 0, acc_bytes, stream);
    attn_kernel<<<512, 256, 0, stream>>>(kbf, vbf, qfrag, attn_raw, colsum,
                                         w_out, it == 3 ? 1 : 0);
    gru_mlp_kernel<<<1024, 64, 0, stream>>>(slots, attn_raw, colsum, wih, whh,
                                            bih, bhh, nrw, nrb, w1, w2, b2);
  }
  normalize_w_kernel<<<(128 * 4096 * 8) / 256, 256, 0, stream>>>(
      w_out, colsum, 128 * 4096 * 8);
}
